// PointNetPP_51075751084309
// MI455X (gfx1250) — matrix operand order repaired and checked by reading, not compiled, not run on hardware
//
#include <hip/hip_runtime.h>

typedef __attribute__((ext_vector_type(16))) _Float16 v16h;
typedef __attribute__((ext_vector_type(8)))  _Float16 h8;
typedef __attribute__((ext_vector_type(8)))  float    v8f;

#define NB 8
#define NPTS 2048

static __host__ __device__ inline int rnd32(int x){ return (x + 31) & ~31; }
static __host__ __device__ inline int rnd64(int x){ return (x + 63) & ~63; }

// ---------------------------------------------------------------------------
// (B,3,N) -> (B,N,3)
__global__ void k_transpose_xyz(const float* __restrict__ x, float* __restrict__ o, long total){
  long e = (long)blockIdx.x*blockDim.x + threadIdx.x; if (e >= total) return;
  int c = (int)(e % 3); long bn = e / 3; int b = (int)(bn / NPTS); int n = (int)(bn % NPTS);
  o[e] = x[((long)b*3 + c)*NPTS + n];
}

// ---------------------------------------------------------------------------
// farthest point sampling, one block per batch, matches jax scan semantics
__global__ void k_fps(const float* __restrict__ xyz, int N, int npoint, int* __restrict__ out){
  __shared__ float sx[2048], sy[2048], sz[2048], sd[2048];
  __shared__ float rmax[256]; __shared__ int rarg[256];
  int b = blockIdx.x, tid = threadIdx.x;
  for (int i = tid; i < N; i += 256){
    sx[i] = xyz[((long)b*N + i)*3 + 0];
    sy[i] = xyz[((long)b*N + i)*3 + 1];
    sz[i] = xyz[((long)b*N + i)*3 + 2];
    sd[i] = 1e10f;
  }
  __syncthreads();
  int cur = 0;
  for (int it = 0; it < npoint; ++it){
    if (tid == 0) out[b*npoint + it] = cur;
    float cx = sx[cur], cy = sy[cur], cz = sz[cur];
    float best = -1.f; int barg = N;
    for (int i = tid; i < N; i += 256){
      float dx = sx[i]-cx, dy = sy[i]-cy, dz = sz[i]-cz;
      float d = dx*dx + dy*dy + dz*dz;
      float nd = fminf(sd[i], d); sd[i] = nd;
      if (nd > best || (nd == best && i < barg)){ best = nd; barg = i; }
    }
    rmax[tid] = best; rarg[tid] = barg;
    __syncthreads();
    for (int off = 128; off > 0; off >>= 1){
      if (tid < off){
        if (rmax[tid+off] > rmax[tid] || (rmax[tid+off] == rmax[tid] && rarg[tid+off] < rarg[tid])){
          rmax[tid] = rmax[tid+off]; rarg[tid] = rarg[tid+off];
        }
      }
      __syncthreads();
    }
    cur = rarg[0];
    __syncthreads();
  }
}

// gather xyz rows by fps index
__global__ void k_gather3(const float* __restrict__ xyz, const int* __restrict__ idx,
                          int N, int S, float* __restrict__ o, long total){
  long e = (long)blockIdx.x*blockDim.x + threadIdx.x; if (e >= total) return;
  int c = (int)(e % 3); long bs = e / 3; int b = (int)(bs / S);
  int gi = idx[bs];
  o[e] = xyz[((long)b*N + gi)*3 + c];
}

// ---------------------------------------------------------------------------
// ball query: first k indices (ascending) within radius, pad with first hit
__global__ void k_ballquery(const float* __restrict__ xyz, const float* __restrict__ nxyz,
                            int N, int S, float r2, int k, int* __restrict__ gidx, long total){
  long e = (long)blockIdx.x*blockDim.x + threadIdx.x; if (e >= total) return;
  int b = (int)(e / S);
  float cx = nxyz[e*3+0], cy = nxyz[e*3+1], cz = nxyz[e*3+2];
  int cnt = 0, first = -1;
  for (int j = 0; j < N; ++j){
    const float* p = &xyz[((long)b*N + j)*3];
    float dx = p[0]-cx, dy = p[1]-cy, dz = p[2]-cz;
    float d = dx*dx + dy*dy + dz*dz;
    if (d <= r2){
      if (first < 0) first = j;
      gidx[e*(long)k + cnt] = j;
      if (++cnt >= k) break;
    }
  }
  if (first < 0) first = 0;
  for (int j = cnt; j < k; ++j) gidx[e*(long)k + j] = first;
}

// ---------------------------------------------------------------------------
// build grouped f16 input: [ points[gidx] (Cp ch) , xyz[gidx]-center (3 ch) , zero pad ]
__global__ void k_pack_group(const float* __restrict__ xyz, const float* __restrict__ pts,
                             int ldP, int Cp, const float* __restrict__ nxyz,
                             const int* __restrict__ gidx, int S, int k, int N, int kpad,
                             _Float16* __restrict__ o, long total){
  long e = (long)blockIdx.x*blockDim.x + threadIdx.x; if (e >= total) return;
  int ch = (int)(e % kpad); long r = e / kpad;
  int j = (int)(r % k); long bs = r / k; int b = (int)(bs / S);
  int gi = gidx[bs*(long)k + j];
  float v = 0.f;
  if (ch < Cp)          v = pts[((long)b*N + gi)*ldP + ch];
  else if (ch < Cp+3){ int c = ch - Cp; v = xyz[((long)b*N + gi)*3 + c] - nxyz[bs*3 + c]; }
  o[e] = (_Float16)v;
}

// SA3 group-all: [ l2_xyz(3) , l2_points(512) , pad to 544 ]
__global__ void k_pack_groupall(const float* __restrict__ l2xyz, const float* __restrict__ l2p,
                                _Float16* __restrict__ o, long total){
  long e = (long)blockIdx.x*blockDim.x + threadIdx.x; if (e >= total) return;
  int ch = (int)(e % 544); long r = e / 544;
  float v = 0.f;
  if (ch < 3)        v = l2xyz[r*3 + ch];
  else if (ch < 515) v = l2p[r*512 + (ch-3)];
  o[e] = (_Float16)v;
}

// FP3 input: [ l2_points(512) , broadcast l3_points(1024) ] = 1536
__global__ void k_pack_fp3(const float* __restrict__ l2p, const float* __restrict__ l3p,
                           _Float16* __restrict__ o, long total){
  long e = (long)blockIdx.x*blockDim.x + threadIdx.x; if (e >= total) return;
  int ch = (int)(e % 1536); long r = e / 1536; int b = (int)(r / 128);
  float v = (ch < 512) ? l2p[r*512 + ch] : l3p[(long)b*1024 + (ch-512)];
  o[e] = (_Float16)v;
}

// FP2 input: [ l1_points(320) , interp(256) ] = 576
__global__ void k_pack_fp2(const float* __restrict__ l1p, const float* __restrict__ itp,
                           _Float16* __restrict__ o, long total){
  long e = (long)blockIdx.x*blockDim.x + threadIdx.x; if (e >= total) return;
  int ch = (int)(e % 576); long r = e / 576;
  float v = (ch < 320) ? l1p[r*320 + ch] : itp[r*256 + (ch-320)];
  o[e] = (_Float16)v;
}

// FP1 input: [ cls(16) , l0_xyz(3) , l0_points(3) , interp(128) , pad to 160 ]
__global__ void k_pack_fp1(const float* __restrict__ cls, const float* __restrict__ l0xyz,
                           const float* __restrict__ itp, _Float16* __restrict__ o, long total){
  long e = (long)blockIdx.x*blockDim.x + threadIdx.x; if (e >= total) return;
  int ch = (int)(e % 160); long r = e / 160; int b = (int)(r / NPTS);
  float v = 0.f;
  if (ch < 16)       v = cls[b*16 + ch];
  else if (ch < 19)  v = l0xyz[r*3 + (ch-16)];
  else if (ch < 22)  v = l0xyz[r*3 + (ch-19)];
  else if (ch < 150) v = itp[r*128 + (ch-22)];
  o[e] = (_Float16)v;
}

// fp32 weights (cout,cin) -> f16 TRANSPOSED K-major: Wk[k][n], k in [0,kpad), n in [0,npad)
__global__ void k_convert_w(const float* __restrict__ W, _Float16* __restrict__ o,
                            int cout, int cin, int kpad, int npad){
  long e = (long)blockIdx.x*blockDim.x + threadIdx.x;
  if (e >= (long)kpad*npad) return;
  int n = (int)(e % npad); int k = (int)(e / npad);
  o[e] = (_Float16)((k < cin && n < cout) ? W[(long)n*cin + k] : 0.f);
}

// ---------------------------------------------------------------------------
// WMMA GEMM: Y = epilogue(X[P x ldX] * Wk[ldX x npad])   (Wk is K-major transposed weights)
// block = 256 threads = 8 waves; block tile 64 rows x 64 cols;
// wave (wm,wn) computes 16 rows x 32 cols via two v_wmma_f32_16x16x32_f16.
// A fragments: two global_load_b128 per lane straight from X (rows 64B aligned).
// B tile (32 k x 64 n) staged in LDS: one global b128 + one ds b128 store per thread,
// fragments read back as 32 16-bit ds loads per lane (16 k rows x 2 col tiles) (pitch 72 halves, conflict-free).
// mode 0: Yh = f16 relu((acc+b)*s+t), zero pad cols [Cout,ldY)
// mode 1: Yf = f32 acc + b (logits)
__global__ __launch_bounds__(256) void k_gemm_wmma(
    const _Float16* __restrict__ X, int ldX,
    const _Float16* __restrict__ Wk, int ldW,
    const float* __restrict__ bias, const float* __restrict__ scale, const float* __restrict__ shift,
    _Float16* __restrict__ Yh, float* __restrict__ Yf, int Cout, int ldY, int mode){
  __shared__ _Float16 Bs[32*72];              // [k][n], pitch 72 halves (144B, 16B aligned)
  const int tid  = threadIdx.x;
  const int lane = tid & 31;
  const int w    = tid >> 5;
  const int wm   = w & 3;                     // row 16-tile
  const int wn   = w >> 2;                    // col 32-tile
  const int g    = lane >> 4;
  const int mr   = lane & 15;
  const long rowbase = (long)blockIdx.x * 64;
  const int  colbase = blockIdx.y * 64;

  // B staging: thread t covers k = t>>3, n = (t&7)*8   (exactly one b128 each)
  const int sk = tid >> 3, sn = (tid & 7) * 8;
  const _Float16* wptr = Wk + (long)sk * ldW + colbase + sn;
  _Float16* bst = &Bs[sk*72 + sn];

  // A fragment source: lane L reads row rowbase+wm*16+mr, 16B runs at K offsets 8g and 16+8g
  const _Float16* aptr = X + (rowbase + wm*16 + mr) * (long)ldX + g*8;

  const _Float16* bfp = &Bs[g*8*72 + wn*32 + mr];

  v8f acc0 = {}, acc1 = {};
  for (int k0 = 0; k0 < ldX; k0 += 32){
    h8 wv  = *(const h8*)(wptr);  wptr += (long)32 * ldW;      // global_load_b128
    h8 alo = *(const h8*)(aptr);                                // global_load_b128
    h8 ahi = *(const h8*)(aptr + 16);                           // global_load_b128
    aptr += 32;
    if (k0 + 32 < ldX)
      __builtin_prefetch(aptr, 0, 1);                           // global_prefetch_b8
    *(h8*)bst = wv;                                             // ds_store_b128
    v16h a = __builtin_shufflevector(alo, ahi, 0,1,2,3,4,5,6,7,8,9,10,11,12,13,14,15);
    __syncthreads();
    v16h b0, b1;
    #pragma unroll
    for (int s = 0; s < 16; ++s){
      int kk = (s < 8) ? s : 8 + s;
      b0[s] = bfp[kk*72];
      b1[s] = bfp[kk*72 + 16];
    }
    acc0 = __builtin_amdgcn_wmma_f32_16x16x32_f16(false, a, false, b0, (short)0, acc0, false, false);
    acc1 = __builtin_amdgcn_wmma_f32_16x16x32_f16(false, a, false, b1, (short)0, acc1, false, false);
    __syncthreads();
  }

  // D fragment: float j -> M = j + 8*(lane/16), N = lane%16
  const long row0 = rowbase + wm*16 + g*8;
  const int c0 = colbase + wn*32 + mr;
  const int c1 = c0 + 16;
  #pragma unroll
  for (int j = 0; j < 8; ++j){
    long row = row0 + j;
    float v0 = acc0[j], v1 = acc1[j];
    if (mode == 0){
      if (c0 < ldY){
        _Float16 o = (_Float16)0.f;
        if (c0 < Cout){ float t = (v0 + bias[c0]) * scale[c0] + shift[c0]; o = (_Float16)(t > 0.f ? t : 0.f); }
        Yh[row*ldY + c0] = o;
      }
      if (c1 < ldY){
        _Float16 o = (_Float16)0.f;
        if (c1 < Cout){ float t = (v1 + bias[c1]) * scale[c1] + shift[c1]; o = (_Float16)(t > 0.f ? t : 0.f); }
        Yh[row*ldY + c1] = o;
      }
    } else {
      if (c0 < Cout) Yf[row*ldY + c0] = v0 + bias[c0];
      if (c1 < Cout) Yf[row*ldY + c1] = v1 + bias[c1];
    }
  }
}

// ---------------------------------------------------------------------------
// max over k grouped points -> f32 stage features at channel offset
__global__ void k_maxpool(const _Float16* __restrict__ Y, int ldY, int k, int C,
                          float* __restrict__ o, int ldO, int coff, long total){
  long e = (long)blockIdx.x*blockDim.x + threadIdx.x; if (e >= total) return;
  int c = (int)(e % C); long bs = e / C;
  float m = -1e30f;
  for (int j = 0; j < k; ++j) m = fmaxf(m, (float)Y[(bs*(long)k + j)*ldY + c]);
  o[bs*ldO + coff + c] = m;
}

// 3 nearest centers + inverse-distance weights
__global__ void k_top3(const float* __restrict__ xyz1, const float* __restrict__ xyz2,
                       int Np, int S, int* __restrict__ oi, float* __restrict__ ow, long total){
  long e = (long)blockIdx.x*blockDim.x + threadIdx.x; if (e >= total) return;
  int b = (int)(e / Np);
  float x = xyz1[e*3+0], y = xyz1[e*3+1], z = xyz1[e*3+2];
  float d0 = 1e30f, d1 = 1e30f, d2 = 1e30f; int i0 = 0, i1 = 0, i2 = 0;
  for (int s = 0; s < S; ++s){
    const float* q = &xyz2[((long)b*S + s)*3];
    float dx = x-q[0], dy = y-q[1], dz = z-q[2];
    float d = dx*dx + dy*dy + dz*dz;
    if (d < d0){ d2=d1; i2=i1; d1=d0; i1=i0; d0=d; i0=s; }
    else if (d < d1){ d2=d1; i2=i1; d1=d; i1=s; }
    else if (d < d2){ d2=d; i2=s; }
  }
  float w0 = 1.f/(d0+1e-8f), w1 = 1.f/(d1+1e-8f), w2 = 1.f/(d2+1e-8f);
  float s3 = w0 + w1 + w2;
  ow[e*3+0] = w0/s3; ow[e*3+1] = w1/s3; ow[e*3+2] = w2/s3;
  oi[e*3+0] = i0; oi[e*3+1] = i1; oi[e*3+2] = i2;
}

__global__ void k_interp(const _Float16* __restrict__ p2, int ldP, const int* __restrict__ ii,
                         const float* __restrict__ ww, int Np, int S, int C,
                         float* __restrict__ o, long total){
  long e = (long)blockIdx.x*blockDim.x + threadIdx.x; if (e >= total) return;
  int c = (int)(e % C); long r = e / C; int b = (int)(r / Np);
  const int* id = &ii[r*3]; const float* wv = &ww[r*3];
  float v = wv[0]*(float)p2[((long)b*S + id[0])*ldP + c]
          + wv[1]*(float)p2[((long)b*S + id[1])*ldP + c]
          + wv[2]*(float)p2[((long)b*S + id[2])*ldP + c];
  o[e] = v;
}

__global__ void k_logsoftmax(const float* __restrict__ L, float* __restrict__ o, long total){
  long e = (long)blockIdx.x*blockDim.x + threadIdx.x; if (e >= total) return;
  const float* p = &L[e*50];
  float m = -1e30f;
  for (int c = 0; c < 50; ++c) m = fmaxf(m, p[c]);
  float s = 0.f;
  for (int c = 0; c < 50; ++c) s += expf(p[c] - m);
  float lg = logf(s);
  for (int c = 0; c < 50; ++c) o[e*50 + c] = p[c] - m - lg;
}

// ---------------------------------------------------------------------------
struct Layer { const float *W,*b,*s,*t; int cout, cin, kpad, npad; _Float16* wh; };

extern "C" void kernel_launch(void* const* d_in, const int* in_sizes, int n_in,
                              void* d_out, int out_size, void* d_ws, size_t ws_size,
                              hipStream_t stream){
  (void)in_sizes; (void)n_in; (void)out_size; (void)ws_size;
  auto F = [&](int i)->const float*{ return (const float*)d_in[i]; };
  const float* xyz_in = F(0);
  const float* cls    = F(1);
  // params flattened: sorted dict keys, list order, layer dicts as W,b,s,t
  int ii = 2;
  const float* conv2W = F(ii++); const float* conv2b = F(ii++);
  Layer fp1[2], fp2[2], fp3[2], head1, sa1[3][3], sa2[2][3], sa3[3], conv2L;
  auto readL = [&](Layer* Ls, const int* ch, int nL){
    for (int l = 0; l < nL; ++l){
      Ls[l].W = F(ii++); Ls[l].b = F(ii++); Ls[l].s = F(ii++); Ls[l].t = F(ii++);
      Ls[l].cin = ch[l]; Ls[l].cout = ch[l+1];
    }
  };
  { int c[3] = {150,128,128};  readL(fp1, c, 2); }
  { int c[3] = {576,256,128};  readL(fp2, c, 2); }
  { int c[3] = {1536,256,256}; readL(fp3, c, 2); }
  { int c[2] = {128,128};      readL(&head1, c, 1); }
  { int c[3][4] = {{6,32,32,64},{6,64,64,128},{6,64,96,128}};
    for (int b = 0; b < 3; ++b) readL(sa1[b], c[b], 3); }
  { int c[2][4] = {{323,128,128,256},{323,128,196,256}};
    for (int b = 0; b < 2; ++b) readL(sa2[b], c[b], 3); }
  { int c[4] = {515,256,512,1024}; readL(sa3, c, 3); }
  conv2L.W = conv2W; conv2L.b = conv2b; conv2L.s = conv2L.t = nullptr;
  conv2L.cin = 128; conv2L.cout = 50;

  // ---- workspace layout (bump allocator) ----
  char* wsp = (char*)d_ws; size_t off = 0;
  auto alloc = [&](size_t bytes)->void*{
    off = (off + 255) & ~(size_t)255; void* p = wsp + off; off += bytes; return p;
  };
  float*     l0xyz = (float*)alloc((size_t)NB*NPTS*3*4);
  int*       fps1  = (int*)  alloc((size_t)NB*512*4);
  int*       fps2  = (int*)  alloc((size_t)NB*128*4);
  float*     l1xyz = (float*)alloc((size_t)NB*512*3*4);
  float*     l2xyz = (float*)alloc((size_t)NB*128*3*4);
  int*       gidx  = (int*)  alloc((size_t)NB*512*128*4);
  _Float16*  Xbuf  = (_Float16*)alloc((size_t)131072*352*2);
  _Float16*  Y0    = (_Float16*)alloc((size_t)524288*128*2);
  _Float16*  Y1    = (_Float16*)alloc((size_t)524288*128*2);
  float*     l1p   = (float*)alloc((size_t)NB*512*320*4);
  float*     l2p   = (float*)alloc((size_t)NB*128*512*4);
  float*     l3p   = (float*)alloc((size_t)NB*1024*4);
  int*       it3i  = (int*)  alloc((size_t)NB*NPTS*3*4);
  float*     it3w  = (float*)alloc((size_t)NB*NPTS*3*4);
  float*     itbuf = (float*)alloc((size_t)NB*NPTS*128*4);
  float*     logit = (float*)alloc((size_t)NB*NPTS*50*4);
  _Float16*  whB   = (_Float16*)alloc((size_t)8<<20);

  auto G = [](long total){ return dim3((unsigned)((total + 255)/256)); };

  // ---- convert all weights to f16, transposed K-major, K pad 32, N pad 64 ----
  Layer* all[26]; int na = 0;
  for (int b = 0; b < 3; ++b) for (int l = 0; l < 3; ++l) all[na++] = &sa1[b][l];
  for (int b = 0; b < 2; ++b) for (int l = 0; l < 3; ++l) all[na++] = &sa2[b][l];
  for (int l = 0; l < 3; ++l) all[na++] = &sa3[l];
  for (int l = 0; l < 2; ++l) all[na++] = &fp3[l];
  for (int l = 0; l < 2; ++l) all[na++] = &fp2[l];
  for (int l = 0; l < 2; ++l) all[na++] = &fp1[l];
  all[na++] = &head1; all[na++] = &conv2L;
  size_t woff = 0;
  for (int i = 0; i < na; ++i){
    Layer* L = all[i];
    L->kpad = rnd32(L->cin);
    L->npad = rnd64(L->cout);
    L->wh = whB + woff; woff += (size_t)L->kpad * L->npad;
    long t = (long)L->kpad * L->npad;
    k_convert_w<<<G(t), 256, 0, stream>>>(L->W, L->wh, L->cout, L->cin, L->kpad, L->npad);
  }

  // ---- MLP chain runner (ping-pong Y0/Y1) ----
  auto run_chain = [&](const _Float16* X0, Layer* Ls, int nL, long P, int& ldOut)->const _Float16*{
    const _Float16* in = X0; int ldIn = Ls[0].kpad;
    _Float16* outs[2] = { Y0, Y1 }; int pp = 0;
    const _Float16* last = nullptr;
    for (int l = 0; l < nL; ++l){
      int ldY = rnd32(Ls[l].cout);
      dim3 grid((unsigned)(P/64), (unsigned)((ldY + 63)/64));
      k_gemm_wmma<<<grid, 256, 0, stream>>>(in, ldIn, Ls[l].wh, Ls[l].npad,
                                            Ls[l].b, Ls[l].s, Ls[l].t,
                                            outs[pp], nullptr, Ls[l].cout, ldY, 0);
      in = outs[pp]; last = in; ldIn = ldY; pp ^= 1;
    }
    ldOut = ldIn; return last;
  };

  // ================= pipeline =================
  k_transpose_xyz<<<G((long)NB*NPTS*3), 256, 0, stream>>>(xyz_in, l0xyz, (long)NB*NPTS*3);

  // ---- SA1 (MSG, 512 centers) ----
  k_fps<<<NB, 256, 0, stream>>>(l0xyz, NPTS, 512, fps1);
  k_gather3<<<G((long)NB*512*3), 256, 0, stream>>>(l0xyz, fps1, NPTS, 512, l1xyz, (long)NB*512*3);
  const float sa1r[3] = {0.1f, 0.2f, 0.4f}; const int sa1k[3] = {32, 64, 128};
  const int   l1off[3] = {0, 64, 192};
  for (int br = 0; br < 3; ++br){
    int k = sa1k[br]; float r2 = sa1r[br]*sa1r[br];
    k_ballquery<<<G((long)NB*512), 256, 0, stream>>>(l0xyz, l1xyz, NPTS, 512, r2, k, gidx, (long)NB*512);
    long P = (long)NB*512*k;
    k_pack_group<<<G(P*32), 256, 0, stream>>>(l0xyz, l0xyz, 3, 3, l1xyz, gidx, 512, k, NPTS, 32, Xbuf, P*32);
    int ldo; const _Float16* fo = run_chain(Xbuf, sa1[br], 3, P, ldo);
    int C = sa1[br][2].cout;
    k_maxpool<<<G((long)NB*512*C), 256, 0, stream>>>(fo, ldo, k, C, l1p, 320, l1off[br], (long)NB*512*C);
  }

  // ---- SA2 (MSG, 128 centers) ----
  k_fps<<<NB, 256, 0, stream>>>(l1xyz, 512, 128, fps2);
  k_gather3<<<G((long)NB*128*3), 256, 0, stream>>>(l1xyz, fps2, 512, 128, l2xyz, (long)NB*128*3);
  const float sa2r[2] = {0.2f, 0.4f}; const int sa2k[2] = {64, 128};
  const int   l2off[2] = {0, 256};
  for (int br = 0; br < 2; ++br){
    int k = sa2k[br]; float r2 = sa2r[br]*sa2r[br];
    k_ballquery<<<G((long)NB*128), 256, 0, stream>>>(l1xyz, l2xyz, 512, 128, r2, k, gidx, (long)NB*128);
    long P = (long)NB*128*k;
    k_pack_group<<<G(P*352), 256, 0, stream>>>(l1xyz, l1p, 320, 320, l2xyz, gidx, 128, k, 512, 352, Xbuf, P*352);
    int ldo; const _Float16* fo = run_chain(Xbuf, sa2[br], 3, P, ldo);
    int C = sa2[br][2].cout;
    k_maxpool<<<G((long)NB*128*C), 256, 0, stream>>>(fo, ldo, k, C, l2p, 512, l2off[br], (long)NB*128*C);
  }

  // ---- SA3 (group all) ----
  k_pack_groupall<<<G((long)NB*128*544), 256, 0, stream>>>(l2xyz, l2p, Xbuf, (long)NB*128*544);
  { int ldo; const _Float16* fo = run_chain(Xbuf, sa3, 3, (long)NB*128, ldo);
    k_maxpool<<<G((long)NB*1024), 256, 0, stream>>>(fo, ldo, 128, 1024, l3p, 1024, 0, (long)NB*1024); }

  // ---- FP3 (broadcast l3 -> 128 pts) ----
  k_pack_fp3<<<G((long)NB*128*1536), 256, 0, stream>>>(l2p, l3p, Xbuf, (long)NB*128*1536);
  int ld3; const _Float16* fo3 = run_chain(Xbuf, fp3, 2, (long)NB*128, ld3);   // (B,128,256) f16 in Y1

  // ---- FP2 (128 -> 512 pts) ----
  k_top3<<<G((long)NB*512), 256, 0, stream>>>(l1xyz, l2xyz, 512, 128, it3i, it3w, (long)NB*512);
  k_interp<<<G((long)NB*512*256), 256, 0, stream>>>(fo3, ld3, it3i, it3w, 512, 128, 256, itbuf, (long)NB*512*256);
  k_pack_fp2<<<G((long)NB*512*576), 256, 0, stream>>>(l1p, itbuf, Xbuf, (long)NB*512*576);
  int ld2; const _Float16* fo2 = run_chain(Xbuf, fp2, 2, (long)NB*512, ld2);   // (B,512,128) f16 in Y1

  // ---- FP1 (512 -> 2048 pts) ----
  k_top3<<<G((long)NB*NPTS), 256, 0, stream>>>(l0xyz, l1xyz, NPTS, 512, it3i, it3w, (long)NB*NPTS);
  k_interp<<<G((long)NB*NPTS*128), 256, 0, stream>>>(fo2, ld2, it3i, it3w, NPTS, 512, 128, itbuf, (long)NB*NPTS*128);
  k_pack_fp1<<<G((long)NB*NPTS*160), 256, 0, stream>>>(cls, l0xyz, itbuf, Xbuf, (long)NB*NPTS*160);
  Layer fp1h[3] = { fp1[0], fp1[1], head1 };
  int ldh; const _Float16* foh = run_chain(Xbuf, fp1h, 3, (long)NB*NPTS, ldh); // (B,2048,128) f16 in Y0

  // ---- conv2 logits (f32) + log_softmax ----
  { dim3 grid((unsigned)((NB*NPTS)/64), 1u);
    k_gemm_wmma<<<grid, 256, 0, stream>>>(foh, ldh, conv2L.wh, conv2L.npad, conv2L.b,
                                          nullptr, nullptr, nullptr, logit, 50, 50, 1); }
  k_logsoftmax<<<G((long)NB*NPTS), 256, 0, stream>>>(logit, (float*)d_out, (long)NB*NPTS);
}
